// semanticFuseModule_82085414961403
// MI455X (gfx1250) — matrix operand order repaired and checked by reading, not compiled, not run on hardware
//
#include <hip/hip_runtime.h>
#include <math.h>

#define NPIX   16384               // H*W
#define BIMG   4
#define MROWS  (BIMG * NPIX)       // 65536 pixel rows
#define LRELU  0.01f
#define EPSLN  1e-5f

typedef __attribute__((ext_vector_type(16))) _Float16 v16h;
typedef __attribute__((ext_vector_type(8)))  float    v8f;
typedef __attribute__((ext_vector_type(2)))  float    v2f;
typedef __attribute__((ext_vector_type(4)))  float    f32x4;
typedef __attribute__((ext_vector_type(4)))  _Float16 f16x4;

// ---------------------------------------------------------------------------
// Stage A: relu(sem) masking + per-class projections -> all_feat (f16),
// plus f16 transpose of x into (p, c) row-major layout.
// ---------------------------------------------------------------------------
__global__ __launch_bounds__(256)
void feat_kernel(const float* __restrict__ x, const float* __restrict__ sem,
                 const float* __restrict__ wa, const float* __restrict__ ba,
                 const float* __restrict__ wb, const float* __restrict__ bb,
                 _Float16* __restrict__ feat, _Float16* __restrict__ xt)
{
    int p = blockIdx.x * blockDim.x + threadIdx.x;  // 0..65535
    int b = p >> 14, n = p & (NPIX - 1);
    const float* xb = x + ((size_t)b * 64) * NPIX + n;   // x[b,c,n], c-stride NPIX

    float acc[64];
    #pragma unroll
    for (int d = 0; d < 64; ++d) acc[d] = 0.f;

    for (int c = 0; c < 64; ++c) {
        float xv = xb[(size_t)c * NPIX];
        xt[(size_t)p * 64 + c] = (_Float16)xv;
        #pragma unroll
        for (int d = 0; d < 4; ++d) acc[d] += xv * wa[c * 4 + d];
        #pragma unroll
        for (int kk = 0; kk < 20; ++kk) {
            #pragma unroll
            for (int d = 0; d < 3; ++d)
                acc[4 + kk * 3 + d] += xv * wb[kk * 192 + c * 3 + d];
        }
    }
    const float* sb = sem + ((size_t)b * 21) * NPIX + n;
    float s0 = fmaxf(sb[0], 0.f);
    #pragma unroll
    for (int d = 0; d < 4; ++d)
        feat[(size_t)p * 64 + d] = (_Float16)(acc[d] * s0 + ba[d]);
    #pragma unroll
    for (int kk = 0; kk < 20; ++kk) {
        float sk = fmaxf(sb[(size_t)(kk + 1) * NPIX], 0.f);
        #pragma unroll
        for (int d = 0; d < 3; ++d)
            feat[(size_t)p * 64 + 4 + kk * 3 + d] =
                (_Float16)(acc[4 + kk * 3 + d] * sk + bb[kk * 3 + d]);
    }
}

// ---------------------------------------------------------------------------
// Pack a weight matrix W[K x NOUT] (f32) into WMMA B-fragment order (f16):
// fragment f = tn*(K/32)+kbi ; per lane one v16h with
//   b[h] = W[(kbi*32 + 16*hs + h) * NOUT + tn*16 + lrow]
// so each GEMM wave fetches its B operand as one 32-byte load.
// ---------------------------------------------------------------------------
template<int K, int NOUT>
__global__ __launch_bounds__(256)
void pack_w_kernel(const float* __restrict__ W, _Float16* __restrict__ BF)
{
    int t = blockIdx.x * blockDim.x + threadIdx.x;
    const int NF = (NOUT / 16) * (K / 32);
    if (t >= NF * 32) return;
    int f = t >> 5, lane = t & 31;
    int tn = f / (K / 32), kbi = f % (K / 32);
    int hs = lane >> 4, lrow = lane & 15;
    const float* wcol = W + (size_t)(kbi * 32 + 16 * hs) * NOUT + tn * 16 + lrow;
    _Float16* dst = BF + ((size_t)f * 32 + lane) * 16;
    #pragma unroll
    for (int h = 0; h < 16; ++h) dst[h] = (_Float16)wcol[(size_t)h * NOUT];
}

// ---------------------------------------------------------------------------
// WMMA GEMM: C[M x NOUT] = A[M x K] (f16) @ W + bias, with W pre-packed into
// B-fragments (WF).  The whole packed blob (<=32KB) is staged into LDS with
// CDNA5 async global->LDS copies (ASYNCcnt) and read back as ds_load_b128.
// EPI: 0 = f16 store, 1 = f32 store, 2 = f32 + f16 residual + NCHW transpose
// ---------------------------------------------------------------------------
template<int K, int NOUT, int EPI>
__global__ __launch_bounds__(256)
void gemm_kernel(const _Float16* __restrict__ A, const _Float16* __restrict__ WF,
                 const float* __restrict__ bias, void* __restrict__ out,
                 const _Float16* __restrict__ resid, int M)
{
    __shared__ __align__(32) _Float16 wlds[K * NOUT];   // packed fragments

    // --- async stage of packed W fragments into LDS (tracked by ASYNCcnt) ---
    {
        uint32_t lbase = (uint32_t)(uintptr_t)(&wlds[0]);
        const char* gsrc = (const char*)WF;
        #pragma unroll 1
        for (uint32_t off = threadIdx.x * 16u; off < (uint32_t)(K * NOUT * 2);
             off += 256u * 16u) {
            uint32_t l = lbase + off;
            uint64_t g = (uint64_t)(uintptr_t)(gsrc + off);
            asm volatile("global_load_async_to_lds_b128 %0, %1, off"
                         :: "v"(l), "v"(g) : "memory");
        }
        asm volatile("s_wait_asynccnt 0x0" ::: "memory");
        __syncthreads();
    }

    const int lane = threadIdx.x & 31;
    const int wave = threadIdx.x >> 5;
    const int tilesN = NOUT / 16;
    int tile = blockIdx.x * (blockDim.x >> 5) + wave;
    int m0   = (tile / tilesN) * 16;
    int tn   = tile % tilesN;
    if (m0 >= M) return;
    int n0 = tn * 16;

    const int hs   = lane >> 4;     // 0: lanes 0-15, 1: lanes 16-31
    const int lrow = lane & 15;
    const v16h* bf_lds = (const v16h*)wlds;

    v8f c = {};
    #pragma unroll
    for (int kbi = 0; kbi < K / 32; ++kbi) {
        // A fragment (contiguous k order, same as packed B): elems 0..15 -> K = kb+16*hs+0..15
        const _Float16* arow = A + (size_t)(m0 + lrow) * K + kbi * 32 + 16 * hs;
        union { v16h v; f32x4 q[2]; } a;
        a.q[0] = *(const f32x4*)(arow);
        a.q[1] = *(const f32x4*)(arow + 8);
        v16h bfr = bf_lds[((size_t)(tn * (K / 32) + kbi)) * 32 + lane];
        c = __builtin_amdgcn_wmma_f32_16x16x32_f16(false, a.v, false, bfr,
                                                   (short)0, c, false, false);
    }

    int   ncol = n0 + lrow;
    float bv   = bias[ncol];
    #pragma unroll
    for (int j = 0; j < 8; ++j) {
        int   row = m0 + j + 8 * hs;           // C/D layout: M = j (+8 for hs)
        float val = c[j] + bv;
        if (EPI == 0) {
            ((_Float16*)out)[(size_t)row * NOUT + ncol] = (_Float16)val;
        } else if (EPI == 1) {
            ((float*)out)[(size_t)row * NOUT + ncol] = val;
        } else {
            val += (float)resid[(size_t)row * 64 + ncol];
            int bimg = row >> 14, n = row & (NPIX - 1);
            ((float*)out)[((size_t)(bimg * 64 + ncol) << 14) + n] = val;
        }
    }
}

// ---------------------------------------------------------------------------
// LayerNorm(128) + leaky_relu, f32 in -> f16 out. One wave per row.
// ---------------------------------------------------------------------------
__global__ __launch_bounds__(256)
void ln_leaky_kernel(const float* __restrict__ in, const float* __restrict__ g,
                     const float* __restrict__ be, _Float16* __restrict__ out, int M)
{
    const int lane = threadIdx.x & 31;
    const int row  = blockIdx.x * 8 + (threadIdx.x >> 5);
    if (row >= M) return;
    f32x4 xv = *(const f32x4*)(in + (size_t)row * 128 + lane * 4);
    float s  = xv[0] + xv[1] + xv[2] + xv[3];
    float s2 = xv[0]*xv[0] + xv[1]*xv[1] + xv[2]*xv[2] + xv[3]*xv[3];
    #pragma unroll
    for (int off = 16; off > 0; off >>= 1) {
        s  += __shfl_xor(s,  off, 32);
        s2 += __shfl_xor(s2, off, 32);
    }
    float mean = s * (1.f / 128.f);
    float var  = s2 * (1.f / 128.f) - mean * mean;
    float inv  = rsqrtf(var + EPSLN);
    f32x4 gv = *(const f32x4*)(g  + lane * 4);
    f32x4 bv = *(const f32x4*)(be + lane * 4);
    f16x4 o;
    #pragma unroll
    for (int i = 0; i < 4; ++i) {
        float t = (xv[i] - mean) * inv * gv[i] + bv[i];
        o[i] = (_Float16)(t > 0.f ? t : LRELU * t);
    }
    *(f16x4*)(out + (size_t)row * 128 + lane * 4) = o;
}

// ---------------------------------------------------------------------------
// Channel-wise MHA core: one block per (b, head). 8 waves accumulate
// S = q k^T (16x16, K=n=16384) with v_wmma_f32_16x16x4_f32 plus per-lane
// sum-of-squares (normalization is separable row/col scaling of S).
// Then softmax(16x16) in LDS and out = attn @ v over all pixels (VALU).
// ---------------------------------------------------------------------------
#define SIDX(d, e) (((d) & 7) * 32 + (e) + (((d) >> 3) << 4))

__global__ __launch_bounds__(256)
void attn_kernel(const float* __restrict__ Q, const float* __restrict__ Km,
                 const float* __restrict__ V, const float* __restrict__ rescale,
                 _Float16* __restrict__ outf)
{
    const int bh = blockIdx.x, b = bh >> 2, h = bh & 3;
    const size_t rowbase = (size_t)b * NPIX;
    const int lane = threadIdx.x & 31, wave = threadIdx.x >> 5;
    const int lrow = lane & 15, hs = lane >> 4;

    __shared__ float lds_s[8 * 256];
    __shared__ float lds_nq[8 * 32], lds_nk[8 * 32];
    __shared__ float attn_s[256];
    __shared__ float nrm_q[16], nrm_k[16];

    v8f  c  = {};
    float nq = 0.f, nk = 0.f;
    for (int n0 = wave * 4; n0 < NPIX; n0 += 32) {
        // A (f32 16x4): lanes 0-15 -> K0,K1 ; lanes 16-31 -> K2,K3
        const float* qp = Q  + (rowbase + n0 + 2 * hs) * 64 + h * 16 + lrow;
        const float* kp = Km + (rowbase + n0 + 2 * hs) * 64 + h * 16 + lrow;
        v2f a;  a[0] = qp[0]; a[1] = qp[64];
        v2f bf; bf[0] = kp[0]; bf[1] = kp[64];
        nq += a[0] * a[0] + a[1] * a[1];
        nk += bf[0] * bf[0] + bf[1] * bf[1];
        c = __builtin_amdgcn_wmma_f32_16x16x4_f32(false, a, false, bf,
                                                  (short)0, c, false, false);
    }
    #pragma unroll
    for (int j = 0; j < 8; ++j) lds_s[wave * 256 + j * 32 + lane] = c[j];
    lds_nq[wave * 32 + lane] = nq;
    lds_nk[wave * 32 + lane] = nk;
    __syncthreads();

    {
        float ssum = 0.f;
        #pragma unroll
        for (int w = 0; w < 8; ++w) ssum += lds_s[w * 256 + threadIdx.x];
        attn_s[threadIdx.x] = ssum;
    }
    if (threadIdx.x < 16) {
        float aq = 0.f, ak = 0.f;
        #pragma unroll
        for (int w = 0; w < 8; ++w) {
            aq += lds_nq[w * 32 + threadIdx.x] + lds_nq[w * 32 + threadIdx.x + 16];
            ak += lds_nk[w * 32 + threadIdx.x] + lds_nk[w * 32 + threadIdx.x + 16];
        }
        nrm_q[threadIdx.x] = sqrtf(aq);
        nrm_k[threadIdx.x] = sqrtf(ak);
    }
    __syncthreads();

    if (threadIdx.x < 16) {                      // softmax per row d
        const int   d  = threadIdx.x;
        const float qn = nrm_q[d] + 1e-8f;
        const float rs = rescale[h];
        float rowv[16], mx = -1e30f;
        #pragma unroll
        for (int e = 0; e < 16; ++e) {
            float v = attn_s[SIDX(d, e)] / (qn * (nrm_k[e] + 1e-8f)) * rs;
            rowv[e] = v; mx = fmaxf(mx, v);
        }
        float sum = 0.f;
        #pragma unroll
        for (int e = 0; e < 16; ++e) { rowv[e] = __expf(rowv[e] - mx); sum += rowv[e]; }
        float isum = 1.f / sum;
        #pragma unroll
        for (int e = 0; e < 16; ++e) attn_s[SIDX(d, e)] = rowv[e] * isum;
    }
    __syncthreads();

    // out[d, n] = sum_e attn[d,e] * v[e,n]; coalesced 64-byte v rows
    for (int n = threadIdx.x; n < NPIX; n += 256) {
        const float* vp = V + (rowbase + n) * 64 + h * 16;
        float vv[16];
        #pragma unroll
        for (int e = 0; e < 16; e += 4) {
            f32x4 t4 = *(const f32x4*)(vp + e);
            vv[e] = t4[0]; vv[e+1] = t4[1]; vv[e+2] = t4[2]; vv[e+3] = t4[3];
        }
        _Float16 o16[16] __attribute__((aligned(16)));
        #pragma unroll
        for (int d = 0; d < 16; ++d) {
            float o = 0.f;
            #pragma unroll
            for (int e = 0; e < 16; ++e) o += attn_s[SIDX(d, e)] * vv[e];
            o16[d] = (_Float16)o;
        }
        f32x4* dp = (f32x4*)(outf + (rowbase + n) * 64 + h * 16);
        dp[0] = ((const f32x4*)o16)[0];
        dp[1] = ((const f32x4*)o16)[1];
    }
}

// ---------------------------------------------------------------------------
extern "C" void kernel_launch(void* const* d_in, const int* in_sizes, int n_in,
                              void* d_out, int out_size, void* d_ws, size_t ws_size,
                              hipStream_t stream)
{
    const float* x    = (const float*)d_in[0];
    const float* sem  = (const float*)d_in[1];
    const float* wa   = (const float*)d_in[2];
    const float* ba   = (const float*)d_in[3];
    const float* wb   = (const float*)d_in[4];
    const float* bb   = (const float*)d_in[5];
    const float* in_w = (const float*)d_in[6];
    const float* in_b = (const float*)d_in[7];
    const float* w1   = (const float*)d_in[8];
    const float* b1   = (const float*)d_in[9];
    const float* g1   = (const float*)d_in[10];
    const float* be1  = (const float*)d_in[11];
    const float* w2   = (const float*)d_in[12];
    const float* b2   = (const float*)d_in[13];
    const float* g2   = (const float*)d_in[14];
    const float* be2  = (const float*)d_in[15];
    const float* w3   = (const float*)d_in[16];
    const float* b3   = (const float*)d_in[17];
    const float* wq   = (const float*)d_in[18];
    const float* bq   = (const float*)d_in[19];
    const float* wk   = (const float*)d_in[20];
    const float* bk   = (const float*)d_in[21];
    const float* wv   = (const float*)d_in[22];
    const float* bv   = (const float*)d_in[23];
    const float* wo   = (const float*)d_in[24];
    const float* bo   = (const float*)d_in[25];
    const float* rsc  = (const float*)d_in[26];

    char* ws = (char*)d_ws;                         // ~80.25 MiB, with reuse
    _Float16* XT   = (_Float16*)(ws + (0ull  << 20));   // 8 MiB
    _Float16* FEAT = (_Float16*)(ws + (8ull  << 20));   // 8 MiB (later: attn out)
    _Float16* YT   = (_Float16*)(ws + (16ull << 20));   // 8 MiB
    float*    H32  = (float*)   (ws + (24ull << 20));   // 32 MiB (later: Q, K)
    _Float16* H16  = (_Float16*)(ws + (56ull << 20));   // 16 MiB (later: V f32)
    _Float16* XM   = (_Float16*)(ws + (72ull << 20));   // 8 MiB
    float*    Qb   = (float*)   (ws + (24ull << 20));
    float*    Kb   = (float*)   (ws + (40ull << 20));
    float*    Vb   = (float*)   (ws + (56ull << 20));
    _Float16* OUTF = (_Float16*)(ws + (8ull  << 20));
    // packed B-fragment slabs (32 KiB each) at +80 MiB
    char* pk = ws + (80ull << 20);
    _Float16* P_INW = (_Float16*)(pk + 0 * 32768);
    _Float16* P_W1  = (_Float16*)(pk + 1 * 32768);
    _Float16* P_W2  = (_Float16*)(pk + 2 * 32768);
    _Float16* P_W3  = (_Float16*)(pk + 3 * 32768);
    _Float16* P_WQ  = (_Float16*)(pk + 4 * 32768);
    _Float16* P_WK  = (_Float16*)(pk + 5 * 32768);
    _Float16* P_WV  = (_Float16*)(pk + 6 * 32768);
    _Float16* P_WO  = (_Float16*)(pk + 7 * 32768);
    float* out = (float*)d_out;

    dim3 blk(256);
    // pack all weight matrices into B-fragment layout (f16)
    pack_w_kernel<64, 64>  <<<1, blk, 0, stream>>>(in_w, P_INW);
    pack_w_kernel<64, 128> <<<2, blk, 0, stream>>>(w1,  P_W1);
    pack_w_kernel<128, 128><<<4, blk, 0, stream>>>(w2,  P_W2);
    pack_w_kernel<128, 64> <<<2, blk, 0, stream>>>(w3,  P_W3);
    pack_w_kernel<64, 64>  <<<1, blk, 0, stream>>>(wq,  P_WQ);
    pack_w_kernel<64, 64>  <<<1, blk, 0, stream>>>(wk,  P_WK);
    pack_w_kernel<64, 64>  <<<1, blk, 0, stream>>>(wv,  P_WV);
    pack_w_kernel<64, 64>  <<<1, blk, 0, stream>>>(wo,  P_WO);

    feat_kernel<<<MROWS / 256, blk, 0, stream>>>(x, sem, wa, ba, wb, bb, FEAT, XT);
    gemm_kernel<64, 64, 0><<<(MROWS / 16) * 4 / 8, blk, 0, stream>>>(FEAT, P_INW, in_b, YT, nullptr, MROWS);
    gemm_kernel<64, 128, 1><<<(MROWS / 16) * 8 / 8, blk, 0, stream>>>(XT, P_W1, b1, H32, nullptr, MROWS);
    ln_leaky_kernel<<<MROWS / 8, blk, 0, stream>>>(H32, g1, be1, H16, MROWS);
    gemm_kernel<128, 128, 1><<<(MROWS / 16) * 8 / 8, blk, 0, stream>>>(H16, P_W2, b2, H32, nullptr, MROWS);
    ln_leaky_kernel<<<MROWS / 8, blk, 0, stream>>>(H32, g2, be2, H16, MROWS);
    gemm_kernel<128, 64, 0><<<(MROWS / 16) * 4 / 8, blk, 0, stream>>>(H16, P_W3, b3, XM, nullptr, MROWS);
    gemm_kernel<64, 64, 1><<<(MROWS / 16) * 4 / 8, blk, 0, stream>>>(XM, P_WQ, bq, Qb, nullptr, MROWS);
    gemm_kernel<64, 64, 1><<<(MROWS / 16) * 4 / 8, blk, 0, stream>>>(YT, P_WK, bk, Kb, nullptr, MROWS);
    gemm_kernel<64, 64, 1><<<(MROWS / 16) * 4 / 8, blk, 0, stream>>>(YT, P_WV, bv, Vb, nullptr, MROWS);
    attn_kernel<<<BIMG * 4, blk, 0, stream>>>(Qb, Kb, Vb, rsc, OUTF);
    gemm_kernel<64, 64, 2><<<(MROWS / 16) * 4 / 8, blk, 0, stream>>>(OUTF, P_WO, bo, out, XM, MROWS);
}
